// MistralAttention_15839839388288
// MI455X (gfx1250) — matrix operand order repaired and checked by reading, not compiled, not run on hardware
//
#include <hip/hip_runtime.h>

typedef _Float16 v16h __attribute__((ext_vector_type(16)));
typedef _Float16 v8h  __attribute__((ext_vector_type(8)));
typedef float    v8f  __attribute__((ext_vector_type(8)));

constexpr int Bc   = 2;
constexpr int SEQ  = 2048;
constexpr int HID  = 4096;
constexpr int NH   = 32;
constexpr int KVH  = 8;
constexpr int HD   = 128;

// ---------------- WMMA wrapper ----------------
__device__ inline v8f wmma_f16(v16h a, v16h b, v8f c) {
  // D = A(16x32 f16) * B(32x16 f16) + C(16x16 f32)
  return __builtin_amdgcn_wmma_f32_16x16x32_f16(
      /*neg_a=*/false, a, /*neg_b=*/false, b,
      /*c_mod=*/(short)0, c, /*reuse_a=*/false, /*reuse_b=*/false);
}

// ---------------- fragment loaders ----------------
// A fragment (16x32, row-major src, ld in elements):
//   lane m = lane&15; half-group hi = lane>>4
//   halves 0..7  = src[m][hi*8 + j],  halves 8..15 = src[m][16 + hi*8 + j]
__device__ inline v16h load_a_frag(const _Float16* base, int ld, int lane) {
  const int m = lane & 15, hi8 = (lane >> 4) << 3;
  const _Float16* p = base + m * ld + hi8;
  v8h a = *(const v8h*)p;
  v8h b = *(const v8h*)(p + 16);
  return __builtin_shufflevector(a, b, 0,1,2,3,4,5,6,7,8,9,10,11,12,13,14,15);
}
__device__ inline v16h load_a_frag(const float* base, int ld, int lane) {
  const int m = lane & 15, hi8 = (lane >> 4) << 3;
  const float* p = base + m * ld + hi8;
  v8f x0 = *(const v8f*)p;
  v8f x1 = *(const v8f*)(p + 16);
  v8h a = __builtin_convertvector(x0, v8h);
  v8h b = __builtin_convertvector(x1, v8h);
  return __builtin_shufflevector(a, b, 0,1,2,3,4,5,6,7,8,9,10,11,12,13,14,15);
}
// B fragment (32x16), loaded column-major: src row n holds the 32 K-values.
//   lane n = lane&15; halves 0..7 = src[n][hi*8 + j], halves 8..15 = src[n][16 + hi*8 + j]
__device__ inline v16h load_b_frag(const _Float16* base, int ld, int lane) {
  const int n = lane & 15, hi8 = (lane >> 4) << 3;
  const _Float16* p = base + n * ld + hi8;
  v8h a = *(const v8h*)p;
  v8h b = *(const v8h*)(p + 16);
  return __builtin_shufflevector(a, b, 0,1,2,3,4,5,6,7,8,9,10,11,12,13,14,15);
}
__device__ inline v16h load_b_frag(const float* base, int ld, int lane) {
  const int n = lane & 15, hi8 = (lane >> 4) << 3;
  const float* p = base + n * ld + hi8;
  v8f x0 = *(const v8f*)p;
  v8f x1 = *(const v8f*)(p + 16);
  v8h a = __builtin_convertvector(x0, v8h);
  v8h b = __builtin_convertvector(x1, v8h);
  return __builtin_shufflevector(a, b, 0,1,2,3,4,5,6,7,8,9,10,11,12,13,14,15);
}

// ---------------- GEMM: C = A(MxK) * W(NxK)^T ----------------
// One wave per 32(M) x 128(N) tile (2 A-tiles reuse each B fragment:
// 8 A-loads + 32 B-loads per 16 WMMAs). Epilogue modes:
//   MODE_ROPE   : apply rotary embedding, store f16 in (B, N/128, S, 128) layout
//   MODE_TRANS  : store f16 transposed (B, N/128, 128, S)  (for V)
//   MODE_OUT_F32: plain fp32 row-major store (final output)
enum { MODE_ROPE = 0, MODE_TRANS = 1, MODE_OUT_F32 = 2 };

template <int MODE, typename TA>
__global__ __launch_bounds__(32)
void gemm_kernel(const TA* __restrict__ A, const float* __restrict__ W,
                 void* __restrict__ Cout, int M, int N, int K,
                 const float* __restrict__ cosp, const float* __restrict__ sinp) {
  const int lane = threadIdx.x;
  const int col0 = blockIdx.x * 128;
  const int row0 = blockIdx.y * 32;

  v8f acc[2][8] = {};
  for (int kk = 0; kk < K; kk += 32) {
    v16h a0 = load_a_frag(A + (long)row0 * K + kk, K, lane);
    v16h a1 = load_a_frag(A + (long)(row0 + 16) * K + kk, K, lane);
#pragma unroll
    for (int t = 0; t < 8; ++t) {
      v16h b = load_b_frag(W + (long)(col0 + t * 16) * K + kk, K, lane);
      acc[0][t] = wmma_f16(a0, b, acc[0][t]);
      acc[1][t] = wmma_f16(a1, b, acc[1][t]);
    }
  }

  // C layout: VGPR r -> row (r + 8*(lane>=16)), col (lane&15) within 16x16 tile
  const int halfoff = (lane >> 4) << 3;
  const int ncol    = lane & 15;

#pragma unroll
  for (int mt = 0; mt < 2; ++mt) {
    const int rbase = row0 + mt * 16;
    if constexpr (MODE == MODE_OUT_F32) {
      float* C = (float*)Cout;
#pragma unroll
      for (int t = 0; t < 8; ++t)
#pragma unroll
        for (int r = 0; r < 8; ++r) {
          const int m = rbase + r + halfoff;
          C[(long)m * N + col0 + t * 16 + ncol] = acc[mt][t][r];
        }
    } else {
      _Float16* C = (_Float16*)Cout;
      const int NHloc = N >> 7;        // heads covered by this projection
      const int h     = col0 >> 7;     // this block is exactly one head
#pragma unroll
      for (int r = 0; r < 8; ++r) {
        const int m    = rbase + r + halfoff;  // row in (B*S)
        const int b    = m / SEQ;
        const int spos = m % SEQ;
        if constexpr (MODE == MODE_ROPE) {
          const long basei = (((long)(b * NHloc + h)) * SEQ + spos) * HD;
#pragma unroll
          for (int t = 0; t < 4; ++t) {
            const int dl = t * 16 + ncol;              // dl in [0,64)
            const float c = cosp[(long)m * HD + dl];   // cos[dl+64]==cos[dl]
            const float s = sinp[(long)m * HD + dl];
            const float lo = acc[mt][t][r], hv = acc[mt][t + 4][r];
            C[basei + dl]      = (_Float16)(lo * c - hv * s);
            C[basei + dl + 64] = (_Float16)(hv * c + lo * s);
          }
        } else {  // MODE_TRANS: (b, h, d, s)
#pragma unroll
          for (int t = 0; t < 8; ++t) {
            const int dl = t * 16 + ncol;
            C[(((long)(b * NHloc + h)) * HD + dl) * SEQ + spos] =
                (_Float16)acc[mt][t][r];
          }
        }
      }
    }
  }
}

// ---------------- Flash attention ----------------
// One wave per (b, h, 16 query rows). Streaming softmax over 32-wide K tiles.
__global__ __launch_bounds__(32)
void flash_kernel(const _Float16* __restrict__ Qh,   // (B, NH, S, D)
                  const _Float16* __restrict__ Kh,   // (B, KVH, S, D)
                  const _Float16* __restrict__ Vt,   // (B, KVH, D, S)
                  _Float16* __restrict__ Oh) {       // (B*S, NH*D)
  const int lane  = threadIdx.x;
  const int qbase = blockIdx.x * 16;
  const int h     = blockIdx.y;
  const int b     = blockIdx.z;
  const int hk    = h >> 2;  // NH/KVH = 4
  const float scaling = 0.08838834764831845f;  // 1/sqrt(128)

  const _Float16* Qp = Qh + ((long)(b * NH + h) * SEQ) * HD;
  const _Float16* Kp = Kh + ((long)(b * KVH + hk) * SEQ) * HD;
  const _Float16* Vp = Vt + ((long)(b * KVH + hk) * HD) * SEQ;

  __shared__ _Float16 Plds[16 * 32];

  v16h aq[4];
#pragma unroll
  for (int c = 0; c < 4; ++c)
    aq[c] = load_a_frag(Qp + (long)qbase * HD + c * 32, HD, lane);

  v8f o[8] = {};
  float mr[8], lr[8];
#pragma unroll
  for (int r = 0; r < 8; ++r) { mr[r] = -1e30f; lr[r] = 0.0f; }

  const int halfoff = (lane >> 4) << 3;
  const int ncol    = lane & 15;

  for (int kb = 0; kb < qbase + 16; kb += 32) {
    // scores: two 16x16 tiles covering k in [kb, kb+32)
    v8f sc0 = {}, sc1 = {};
#pragma unroll
    for (int c = 0; c < 4; ++c) {
      v16h bk0 = load_b_frag(Kp + (long)kb * HD + c * 32, HD, lane);
      v16h bk1 = load_b_frag(Kp + (long)(kb + 16) * HD + c * 32, HD, lane);
      sc0 = wmma_f16(aq[c], bk0, sc0);
      sc1 = wmma_f16(aq[c], bk1, sc1);
    }
    // streaming softmax per row (row = r + 8*(lane>=16); 16-lane reductions)
#pragma unroll
    for (int r = 0; r < 8; ++r) {
      const int q = qbase + r + halfoff;
      float v0 = sc0[r] * scaling;
      float v1 = sc1[r] * scaling;
      if (kb + ncol > q)      v0 = -1e30f;  // causal mask
      if (kb + 16 + ncol > q) v1 = -1e30f;
      float mx = fmaxf(v0, v1);
#pragma unroll
      for (int off = 1; off < 16; off <<= 1)
        mx = fmaxf(mx, __shfl_xor(mx, off, 32));
      const float mnew = fmaxf(mr[r], mx);
      const float corr = __expf(mr[r] - mnew);
      const float p0 = __expf(v0 - mnew);
      const float p1 = __expf(v1 - mnew);
      float rs = p0 + p1;
#pragma unroll
      for (int off = 1; off < 16; off <<= 1)
        rs += __shfl_xor(rs, off, 32);
      lr[r] = lr[r] * corr + rs;
      mr[r] = mnew;
#pragma unroll
      for (int t = 0; t < 8; ++t) o[t][r] *= corr;
      const int m = r + halfoff;
      Plds[m * 32 + ncol]      = (_Float16)p0;   // reshape C-layout -> A-layout
      Plds[m * 32 + 16 + ncol] = (_Float16)p1;
    }
    __syncthreads();
    v16h ap = load_a_frag((const _Float16*)&Plds[0], 32, lane);
#pragma unroll
    for (int t = 0; t < 8; ++t) {  // o(16x128) += P(16x32) * V(32x128)
      v16h bv = load_b_frag(Vp + (long)(t * 16) * SEQ + kb, SEQ, lane);
      o[t] = wmma_f16(ap, bv, o[t]);
    }
    __syncthreads();
  }

  // normalize + store f16 in (B*S, NH*D) for the O-projection
#pragma unroll
  for (int r = 0; r < 8; ++r) {
    const float inv = 1.0f / lr[r];
    const int m = qbase + r + halfoff;
    const long basei = ((long)(b * SEQ + m)) * (NH * HD) + (long)h * HD;
#pragma unroll
    for (int t = 0; t < 8; ++t)
      Oh[basei + t * 16 + ncol] = (_Float16)(o[t][r] * inv);
  }
}

extern "C" void kernel_launch(void* const* d_in, const int* in_sizes, int n_in,
                              void* d_out, int out_size, void* d_ws, size_t ws_size,
                              hipStream_t stream) {
  const float* x    = (const float*)d_in[0];
  const float* cosp = (const float*)d_in[1];
  const float* sinp = (const float*)d_in[2];
  const float* wq   = (const float*)d_in[3];
  const float* wk   = (const float*)d_in[4];
  const float* wv   = (const float*)d_in[5];
  const float* wo   = (const float*)d_in[6];
  float* out = (float*)d_out;

  char* ws = (char*)d_ws;
  size_t off = 0;
  auto alloc = [&](size_t bytes) {
    char* p = ws + off;
    off += (bytes + 255) & ~(size_t)255;
    return p;
  };
  _Float16* Qh    = (_Float16*)alloc((size_t)Bc * NH  * SEQ * HD * 2);  // rope'd Q (b,h,s,d)
  _Float16* Khb   = (_Float16*)alloc((size_t)Bc * KVH * SEQ * HD * 2);  // rope'd K (b,hk,s,d)
  _Float16* Vtb   = (_Float16*)alloc((size_t)Bc * KVH * HD * SEQ * 2);  // V^T    (b,hk,d,s)
  _Float16* AttnH = (_Float16*)alloc((size_t)Bc * SEQ * NH * HD * 2);   // attn out f16

  const int M = Bc * SEQ;  // 4096
  dim3 blk(32);

  gemm_kernel<MODE_ROPE, float>
      <<<dim3((NH * HD) / 128, M / 32), blk, 0, stream>>>(x, wq, Qh, M, NH * HD, HID, cosp, sinp);
  gemm_kernel<MODE_ROPE, float>
      <<<dim3((KVH * HD) / 128, M / 32), blk, 0, stream>>>(x, wk, Khb, M, KVH * HD, HID, cosp, sinp);
  gemm_kernel<MODE_TRANS, float>
      <<<dim3((KVH * HD) / 128, M / 32), blk, 0, stream>>>(x, wv, Vtb, M, KVH * HD, HID, nullptr, nullptr);
  flash_kernel<<<dim3(SEQ / 16, NH, Bc), blk, 0, stream>>>(Qh, Khb, Vtb, AttnH);
  gemm_kernel<MODE_OUT_F32, _Float16>
      <<<dim3(HID / 128, M / 32), blk, 0, stream>>>(AttnH, wo, out, M, HID, NH * HD, nullptr, nullptr);
}
